// TreeEnergyLoss_68925635166493
// MI455X (gfx1250) — matrix operand order repaired and checked by reading, not compiled, not run on hardware
//
#include <hip/hip_runtime.h>
#include <cstdint>
#include <cstddef>

// ---------------- problem constants ----------------
#define Bb     4
#define CCc    21
#define Hh     96
#define Ww     96
#define Nn     (Hh*Ww)        // 9216
#define NP1    (Nn+1)         // 9217
#define CLOW   3
#define CHIGH  512
#define EH     (Hh*(Ww-1))    // 9120
#define EV     ((Hh-1)*Ww)    // 9120
#define Ee     (EH+EV)        // 18240
#define EPAD   32768
#define SIGMA_LOW 0.02f

typedef __attribute__((ext_vector_type(8)))  float  v8f;
typedef __attribute__((ext_vector_type(8)))  __bf16 v8bf;
typedef __attribute__((ext_vector_type(16))) __bf16 v16bf;

__device__ __forceinline__ void decode_edge(int e, int& a, int& b) {
    if (e < EH) { int r = e / (Ww-1), c = e % (Ww-1); a = r*Ww + c; b = a + 1; }
    else        { int e2 = e - EH; int r = e2 / Ww, c = e2 % Ww; a = r*Ww + c; b = a + Ww; }
}

// ---------------- elementwise: sigmoid ----------------
__global__ void k_sigmoid(const float* __restrict__ x, float* __restrict__ y, int n) {
    int i = blockIdx.x*blockDim.x + threadIdx.x;
    if (i < n) y[i] = 1.0f / (1.0f + __expf(-x[i]));
}

// ---------------- transpose high feats [B,512,N] -> bf16 [B,N+1,512] ----------------
__global__ void k_transpose_hf(const float* __restrict__ hf, __bf16* __restrict__ hft) {
    __shared__ float tile[32][33];
    int b = blockIdx.z;
    int n0 = blockIdx.x * 32;
    int c0 = blockIdx.y * 32;
    int tx = threadIdx.x, ty = threadIdx.y;  // 32 x 8
    const float* hfb = hf + (size_t)b * CHIGH * Nn;
    #pragma unroll
    for (int r = 0; r < 32; r += 8)
        tile[ty + r][tx] = hfb[(size_t)(c0 + ty + r) * Nn + (n0 + tx)];
    __syncthreads();
    __bf16* out = hft + (size_t)b * NP1 * CHIGH;
    #pragma unroll
    for (int r = 0; r < 32; r += 8)
        out[(size_t)(n0 + ty + r) * CHIGH + (c0 + tx)] = (__bf16)tile[tx][ty + r];
}

__global__ void k_zero_dummy_row(__bf16* __restrict__ hft) {
    int i = blockIdx.x*blockDim.x + threadIdx.x;   // B*CHIGH
    if (i < Bb*CHIGH) {
        int b = i / CHIGH, c = i % CHIGH;
        hft[(size_t)b * NP1 * CHIGH + (size_t)Nn * CHIGH + c] = (__bf16)0.0f;
    }
}

// ---------------- low-dim edge weights -> sort keys ----------------
__global__ void k_edge_low(const float* __restrict__ lf, uint64_t* __restrict__ keys) {
    int idx = blockIdx.x*blockDim.x + threadIdx.x;     // B*EPAD
    if (idx >= Bb*EPAD) return;
    int b = idx / EPAD, e = idx % EPAD;
    uint64_t key;
    if (e < Ee) {
        int a, c; decode_edge(e, a, c);
        const float* base = lf + (size_t)b * CLOW * Nn;
        float d2 = 0.f;
        #pragma unroll
        for (int ch = 0; ch < CLOW; ++ch) {
            float d = base[(size_t)ch*Nn + a] - base[(size_t)ch*Nn + c];
            d2 += d * d;
        }
        key = ((uint64_t)__float_as_uint(d2) << 32) | (uint32_t)e;
    } else key = ~0ull;
    keys[idx] = key;
}

// ---------------- WMMA 16-pair squared distance over 512 bf16 channels ----------------
// One wave computes ||x_pa[m] - x_pb[m]||^2 for m = lane&15 via diagonal of diff-Gram.
__device__ float wmma_dist16(const __bf16* __restrict__ hftb, int pa, int pb,
                             float (*sh)[8], int lane) {
    const bool hi = lane >= 16;
    const __bf16* rA = hftb + (size_t)pa * CHIGH;
    const __bf16* rB = hftb + (size_t)pb * CHIGH;
    v8f acc = {0.f,0.f,0.f,0.f,0.f,0.f,0.f,0.f};
    for (int c = 0; c < CHIGH; c += 32) {
        if (c + 32 < CHIGH) {
            __builtin_prefetch(rA + c + 32, 0, 3);
            __builtin_prefetch(rB + c + 32, 0, 3);
        }
        v8bf dv[4];
        #pragma unroll
        for (int q = 0; q < 4; ++q) {
            v8bf ra = *(const v8bf*)(rA + c + q*8);
            v8bf rb = *(const v8bf*)(rB + c + q*8);
            v8bf d;
            #pragma unroll
            for (int j = 0; j < 8; ++j) d[j] = (__bf16)((float)ra[j] - (float)rb[j]);
            dv[q] = d;
        }
        // A-matrix 16x32 bf16 layout: lanes<16 hold K {0..7,16..23}, lanes>=16 {8..15,24..31}
        v8bf alo = hi ? dv[1] : dv[0];
        v8bf ahi = hi ? dv[3] : dv[2];
        // B-matrix 32x16 layout: lanes<16 hold K {0..7,16..23}, lanes>=16 {8..15,24..31}
        v8bf blo = hi ? dv[1] : dv[0];
        v8bf bhi = hi ? dv[3] : dv[2];
        v16bf af = __builtin_shufflevector(alo, ahi, 0,1,2,3,4,5,6,7,8,9,10,11,12,13,14,15);
        v16bf bf = __builtin_shufflevector(blo, bhi, 0,1,2,3,4,5,6,7,8,9,10,11,12,13,14,15);
        acc = __builtin_amdgcn_wmma_f32_16x16x32_bf16(false, af, false, bf,
                                                      (short)0, acc, false, false);
    }
    #pragma unroll
    for (int r = 0; r < 8; ++r) sh[lane][r] = acc[r];
    __syncthreads();
    int m = lane & 15;
    // C/D layout: C[m][m] lives at (vgpr m, lane m) for m<8, (vgpr m-8, lane m+16) for m>=8
    return (m < 8) ? sh[m][m] : sh[m + 16][m - 8];
}

__global__ void k_edge_high_wmma(const __bf16* __restrict__ hft, uint64_t* __restrict__ keys) {
    const int GRP = EPAD / 16;                       // 2048 groups per batch
    int b = blockIdx.x / GRP, g = blockIdx.x % GRP;
    int lane = threadIdx.x, m = lane & 15;
    int p = g * 16 + m;
    int pa = Nn, pb = Nn;
    if (p < Ee) decode_edge(p, pa, pb);
    const __bf16* hftb = hft + (size_t)b * NP1 * CHIGH;
    __shared__ float sh[32][8];
    float d2 = wmma_dist16(hftb, pa, pb, sh, lane);
    if (lane < 16) {
        uint64_t key = (p < Ee) ? (((uint64_t)__float_as_uint(d2) << 32) | (uint32_t)p)
                                : ~0ull;
        keys[(size_t)b * EPAD + p] = key;
    }
}

__global__ void k_node_w_high_wmma(const __bf16* __restrict__ hft,
                                   const int* __restrict__ parent,
                                   float* __restrict__ w) {
    const int GRP = (NP1 + 15) / 16;                 // 577 groups per batch
    int b = blockIdx.x / GRP, g = blockIdx.x % GRP;
    int lane = threadIdx.x, m = lane & 15;
    int p = g * 16 + m;
    int pc = (p <= Nn) ? p : Nn;
    int par = parent[(size_t)b * NP1 + pc];
    const __bf16* hftb = hft + (size_t)b * NP1 * CHIGH;
    __shared__ float sh[32][8];
    float d2 = wmma_dist16(hftb, pc, par, sh, lane);
    if (lane < 16 && p <= Nn) {
        float wv = (p == 0 || p == Nn) ? 0.f : __expf(-d2);   // sigma = 1.0
        w[(size_t)b * NP1 + p] = wv;
    }
}

// ---------------- low-dim parent-edge weights ----------------
__global__ void k_node_w_low(const float* __restrict__ lf, const int* __restrict__ parent,
                             float* __restrict__ w) {
    int idx = blockIdx.x*blockDim.x + threadIdx.x;   // B*NP1
    if (idx >= Bb*NP1) return;
    int b = idx / NP1, i = idx % NP1;
    int p = parent[idx];
    const float* base = lf + (size_t)b * CLOW * Nn;
    float d2 = 0.f;
    #pragma unroll
    for (int ch = 0; ch < CLOW; ++ch) {
        float a = (i < Nn) ? base[(size_t)ch*Nn + i] : 0.f;
        float c = (p < Nn) ? base[(size_t)ch*Nn + p] : 0.f;
        float d = a - c; d2 += d * d;
    }
    w[idx] = (i == 0 || i == Nn) ? 0.f : __expf(-d2 / SIGMA_LOW);
}

// ---------------- bitonic sort (one workgroup per batch, in-place, global) ----------------
__global__ void k_bitonic_sort(uint64_t* __restrict__ keys) {
    uint64_t* kb = keys + (size_t)blockIdx.x * EPAD;
    for (unsigned k = 2; k <= EPAD; k <<= 1) {
        for (unsigned j = k >> 1; j > 0; j >>= 1) {
            for (unsigned t = threadIdx.x; t < EPAD; t += blockDim.x) {
                unsigned ixj = t ^ j;
                if (ixj > t) {
                    uint64_t a = kb[t], c = kb[ixj];
                    bool up = ((t & k) == 0);
                    if ((a > c) == up) { kb[t] = c; kb[ixj] = a; }
                }
            }
            __syncthreads();
        }
    }
}

// ---------------- serial Kruskal + BFS schedule (lane 0 of each block) ----------------
__global__ void k_mst_bfs(const uint64_t* __restrict__ keys,
                          int* __restrict__ parent, int* __restrict__ order,
                          int* __restrict__ childStart, int* __restrict__ childCount,
                          int* __restrict__ levelStart, int* __restrict__ numLevels,
                          int* __restrict__ uf, int* __restrict__ adj,
                          int* __restrict__ adjCnt, int* __restrict__ depth) {
    if (threadIdx.x != 0) return;
    int b = blockIdx.x;
    const uint64_t* kb = keys + (size_t)b * EPAD;
    int* ufb  = uf      + (size_t)b * Nn;
    int* adjb = adj     + (size_t)b * Nn * 4;
    int* acb  = adjCnt  + (size_t)b * Nn;
    int* parb = parent  + (size_t)b * NP1;
    int* ordb = order   + (size_t)b * Nn;
    int* csb  = childStart + (size_t)b * Nn;
    int* ccb  = childCount + (size_t)b * Nn;
    int* lsb  = levelStart + (size_t)b * (Nn + 2);
    int* depb = depth   + (size_t)b * Nn;
    for (int i = 0; i < Nn; ++i) { ufb[i] = i; acb[i] = 0; depb[i] = -1; }
    int cnt = 0;
    for (int k = 0; k < EPAD && cnt < Nn - 1; ++k) {
        uint32_t e = (uint32_t)(kb[k] & 0xffffffffu);
        if (e >= (uint32_t)Ee) continue;
        int a, c; decode_edge((int)e, a, c);
        int ra = a; while (ufb[ra] != ra) { ufb[ra] = ufb[ufb[ra]]; ra = ufb[ra]; }
        int rc = c; while (ufb[rc] != rc) { ufb[rc] = ufb[ufb[rc]]; rc = ufb[rc]; }
        if (ra != rc) {
            ufb[ra] = rc;
            adjb[a*4 + acb[a]++] = c;
            adjb[c*4 + acb[c]++] = a;
            ++cnt;
        }
    }
    // BFS from node 0; children of a node occupy a contiguous run of order[]
    parb[Nn] = Nn; parb[0] = 0;
    depb[0] = 0; ordb[0] = 0;
    int qt = 1;
    for (int qh = 0; qh < qt; ++qh) {
        int u = ordb[qh];
        csb[u] = qt; int c0 = qt;
        for (int j = 0; j < acb[u]; ++j) {
            int v = adjb[u*4 + j];
            if (depb[v] < 0) { depb[v] = depb[u] + 1; parb[v] = u; ordb[qt++] = v; }
        }
        ccb[u] = qt - c0;
    }
    int nl = 1; lsb[0] = 0;
    for (int k = 1; k < qt; ++k)
        if (depb[ordb[k]] != depb[ordb[k-1]]) lsb[nl++] = k;
    lsb[nl] = qt;
    numLevels[b] = nl;
}

// ---------------- tree filter: up (gather children) + down (read parent) ----------------
__global__ void k_tree_filter(const float* __restrict__ fin, const float* __restrict__ w,
                              const int* __restrict__ parent, const int* __restrict__ order,
                              const int* __restrict__ childStart, const int* __restrict__ childCount,
                              const int* __restrict__ levelStart, const int* __restrict__ numLevels,
                              float* __restrict__ fa, float* __restrict__ fout) {
    const int CH = CCc + 1;
    int b = blockIdx.x;
    const float* finb = fin + (size_t)b * CCc * Nn;
    const float* wb   = w   + (size_t)b * NP1;
    const int* parb = parent + (size_t)b * NP1;
    const int* ordb = order  + (size_t)b * Nn;
    const int* csb  = childStart + (size_t)b * Nn;
    const int* ccb  = childCount + (size_t)b * Nn;
    const int* lsb  = levelStart + (size_t)b * (Nn + 2);
    int nL = numLevels[b];
    float* fab = fa + (size_t)b * CH * NP1;
    for (int t = threadIdx.x; t < CH * NP1; t += blockDim.x) {
        int ch = t / NP1, i = t % NP1;
        float v = (i == Nn) ? 0.f : ((ch < CCc) ? finb[(size_t)ch*Nn + i] : 1.0f);
        fab[t] = v;
    }
    __syncthreads();
    for (int d = nL - 1; d >= 0; --d) {          // leaf -> root
        int s0 = lsb[d], s1 = lsb[d+1], cnt = (s1 - s0) * CH;
        for (int t = threadIdx.x; t < cnt; t += blockDim.x) {
            int k = s0 + t / CH, ch = t % CH;
            int u = ordb[k];
            float acc = fab[(size_t)ch*NP1 + u];
            int cs = csb[u], cc = ccb[u];
            for (int j = 0; j < cc; ++j) {
                int c = ordb[cs + j];
                acc += wb[c] * fab[(size_t)ch*NP1 + c];
            }
            fab[(size_t)ch*NP1 + u] = acc;
        }
        __syncthreads();
    }
    for (int d = 1; d < nL; ++d) {               // root -> leaf (in place)
        int s0 = lsb[d], s1 = lsb[d+1], cnt = (s1 - s0) * CH;
        for (int t = threadIdx.x; t < cnt; t += blockDim.x) {
            int k = s0 + t / CH, ch = t % CH;
            int u = ordb[k], p = parb[u];
            float wl = wb[u];
            float au = fab[(size_t)ch*NP1 + u];
            float sp = fab[(size_t)ch*NP1 + p];
            fab[(size_t)ch*NP1 + u] = au + wl * (sp - wl * au);
        }
        __syncthreads();
    }
    float* foutb = fout + (size_t)b * CCc * Nn;
    for (int t = threadIdx.x; t < CCc * Nn; t += blockDim.x) {
        int ch = t / Nn, i = t % Nn;
        foutb[t] = fab[(size_t)ch*NP1 + i] / fab[(size_t)CCc*NP1 + i];
    }
}

// ---------------- loss reduction (deterministic) ----------------
__global__ void k_loss_partial(const float* __restrict__ prob, const float* __restrict__ as2,
                               const float* __restrict__ roi, float* __restrict__ partials) {
    __shared__ float sA[256], sB[256];
    float accL = 0.f, accR = 0.f;
    int gsz = gridDim.x * blockDim.x;
    int gid = blockIdx.x * blockDim.x + threadIdx.x;
    for (int t = gid; t < Bb*CCc*Nn; t += gsz) {
        int b = t / (CCc*Nn);
        int i = t % Nn;
        int yi = i / Ww, xi = i % Ww;
        float r = roi[(size_t)b*(2*Hh)*(2*Ww) + (size_t)(2*yi)*(2*Ww) + 2*xi];
        accL += r * fabsf(prob[t] - as2[t]);
    }
    for (int t = gid; t < Bb*Nn; t += gsz) {
        int b = t / Nn, i = t % Nn;
        int yi = i / Ww, xi = i % Ww;
        accR += roi[(size_t)b*(2*Hh)*(2*Ww) + (size_t)(2*yi)*(2*Ww) + 2*xi];
    }
    sA[threadIdx.x] = accL; sB[threadIdx.x] = accR;
    __syncthreads();
    for (int s = 128; s > 0; s >>= 1) {
        if (threadIdx.x < s) { sA[threadIdx.x] += sA[threadIdx.x+s]; sB[threadIdx.x] += sB[threadIdx.x+s]; }
        __syncthreads();
    }
    if (threadIdx.x == 0) { partials[blockIdx.x*2] = sA[0]; partials[blockIdx.x*2+1] = sB[0]; }
}

__global__ void k_loss_final(const float* __restrict__ partials, int nblk, float* __restrict__ out) {
    __shared__ float sA[256], sB[256];
    float aL = 0.f, aR = 0.f;
    for (int i = threadIdx.x; i < nblk; i += 256) { aL += partials[2*i]; aR += partials[2*i+1]; }
    sA[threadIdx.x] = aL; sB[threadIdx.x] = aR;
    __syncthreads();
    for (int s = 128; s > 0; s >>= 1) {
        if (threadIdx.x < s) { sA[threadIdx.x] += sA[threadIdx.x+s]; sB[threadIdx.x] += sB[threadIdx.x+s]; }
        __syncthreads();
    }
    if (threadIdx.x == 0) out[0] = (sB[0] > 0.f) ? (sA[0] / sB[0]) : 0.f;
}

// ---------------- host-side launcher ----------------
extern "C" void kernel_launch(void* const* d_in, const int* in_sizes, int n_in,
                              void* d_out, int out_size, void* d_ws, size_t ws_size,
                              hipStream_t stream) {
    const float* preds = (const float*)d_in[0];  // [B,21,96,96]
    const float* lf    = (const float*)d_in[1];  // [B,3,96,96]
    const float* hf    = (const float*)d_in[2];  // [B,512,96,96]
    const float* roi   = (const float*)d_in[3];  // [B,1,192,192]

    char* wp = (char*)d_ws;
    auto alloc = [&](size_t bytes) -> void* {
        void* r = (void*)wp;
        wp += (bytes + 255) & ~(size_t)255;
        return r;
    };
    float*    prob  = (float*)alloc((size_t)Bb*CCc*Nn*4);
    float*    AS1   = (float*)alloc((size_t)Bb*CCc*Nn*4);
    float*    AS2   = (float*)alloc((size_t)Bb*CCc*Nn*4);
    __bf16*   hft   = (__bf16*)alloc((size_t)Bb*NP1*CHIGH*2);
    uint64_t* keysL = (uint64_t*)alloc((size_t)Bb*EPAD*8);
    uint64_t* keysH = (uint64_t*)alloc((size_t)Bb*EPAD*8);
    // tree L
    int* parL = (int*)alloc((size_t)Bb*NP1*4);
    int* ordL = (int*)alloc((size_t)Bb*Nn*4);
    int* csL  = (int*)alloc((size_t)Bb*Nn*4);
    int* ccL  = (int*)alloc((size_t)Bb*Nn*4);
    int* lsL  = (int*)alloc((size_t)Bb*(Nn+2)*4);
    int* nlL  = (int*)alloc((size_t)Bb*4);
    float* wL = (float*)alloc((size_t)Bb*NP1*4);
    // tree H
    int* parH = (int*)alloc((size_t)Bb*NP1*4);
    int* ordH = (int*)alloc((size_t)Bb*Nn*4);
    int* csH  = (int*)alloc((size_t)Bb*Nn*4);
    int* ccH  = (int*)alloc((size_t)Bb*Nn*4);
    int* lsH  = (int*)alloc((size_t)Bb*(Nn+2)*4);
    int* nlH  = (int*)alloc((size_t)Bb*4);
    float* wH = (float*)alloc((size_t)Bb*NP1*4);
    // temps (reused by both MST builds, sequential)
    int* uf     = (int*)alloc((size_t)Bb*Nn*4);
    int* adj    = (int*)alloc((size_t)Bb*Nn*4*4);
    int* adjCnt = (int*)alloc((size_t)Bb*Nn*4);
    int* depth  = (int*)alloc((size_t)Bb*Nn*4);
    float* fa   = (float*)alloc((size_t)Bb*(CCc+1)*NP1*4);
    float* partials = (float*)alloc((size_t)256*2*4);

    // 1. sigmoid
    {
        int n = Bb*CCc*Nn;
        k_sigmoid<<<(n + 255)/256, 256, 0, stream>>>(preds, prob, n);
    }
    // 2. transpose high feats to node-major bf16 + zero dummy row
    k_transpose_hf<<<dim3(Nn/32, CHIGH/32, Bb), dim3(32, 8), 0, stream>>>(hf, hft);
    k_zero_dummy_row<<<(Bb*CHIGH + 255)/256, 256, 0, stream>>>(hft);
    // 3. edge weight keys
    k_edge_low<<<(Bb*EPAD + 255)/256, 256, 0, stream>>>(lf, keysL);
    k_edge_high_wmma<<<Bb*(EPAD/16), 32, 0, stream>>>(hft, keysH);
    // 4. stable sorts (key = weight_bits<<32 | edge_idx)
    k_bitonic_sort<<<Bb, 1024, 0, stream>>>(keysL);
    k_bitonic_sort<<<Bb, 1024, 0, stream>>>(keysH);
    // 5. Kruskal + BFS schedule
    k_mst_bfs<<<Bb, 32, 0, stream>>>(keysL, parL, ordL, csL, ccL, lsL, nlL, uf, adj, adjCnt, depth);
    k_mst_bfs<<<Bb, 32, 0, stream>>>(keysH, parH, ordH, csH, ccH, lsH, nlH, uf, adj, adjCnt, depth);
    // 6. parent-edge weights
    k_node_w_low<<<(Bb*NP1 + 255)/256, 256, 0, stream>>>(lf, parL, wL);
    k_node_w_high_wmma<<<Bb*((NP1 + 15)/16), 32, 0, stream>>>(hft, parH, wH);
    // 7. two tree-filter passes
    k_tree_filter<<<Bb, 1024, 0, stream>>>(prob, wL, parL, ordL, csL, ccL, lsL, nlL, fa, AS1);
    k_tree_filter<<<Bb, 1024, 0, stream>>>(AS1,  wH, parH, ordH, csH, ccH, lsH, nlH, fa, AS2);
    // 8. loss
    k_loss_partial<<<256, 256, 0, stream>>>(prob, AS2, roi, partials);
    k_loss_final<<<1, 256, 0, stream>>>(partials, 256, (float*)d_out);
}
